// LSTM_29875792511463
// MI455X (gfx1250) — matrix operand order repaired and checked by reading, not compiled, not run on hardware
//
#include <hip/hip_runtime.h>

typedef __attribute__((ext_vector_type(16))) __bf16 v16bf;
typedef __attribute__((ext_vector_type(8)))  __bf16 v8bf;
typedef __attribute__((ext_vector_type(8)))  float  v8f;

#define B_    32
#define S_    512
#define D_    1024
#define H_    1024
#define KTOT  2048         // fused K: [x_t | h_{t-1}]
#define KHALF 1024         // K slice per wave-group
#define NWG   64           // workgroups per layer kernel (16 h-cols each)
#define NTH   256          // 8 wave32s per WG: gate = wave&3, khalf = wave>>2
#define XPITCH 1032        // LDS row pitch in bf16 elems (1024 + 8 pad)

// LDS byte offsets (dynamic shared starts at LDS offset 0)
#define XBUF_B   (32 * XPITCH * 2)         // 66048 per x buffer
#define SX0_OFF  0u
#define SX1_OFF  ((unsigned)XBUF_B)
#define SH_OFF   ((unsigned)(2 * XBUF_B))
#define GBUF_OFF ((unsigned)(3 * XBUF_B))
#define SHMEM_B  (3 * XBUF_B + 2 * 4 * 32 * 16 * 4)   // + gbuf 16 KB

__device__ __forceinline__ unsigned short f32_to_bf16(float f) {
    unsigned u = __float_as_uint(f);
    u += 0x7FFFu + ((u >> 16) & 1u);       // round-to-nearest-even
    return (unsigned short)(u >> 16);
}

// CDNA5 async global->LDS copy: 16 B per lane, tracked by ASYNCcnt.
__device__ __forceinline__ void async_load_b128(unsigned lds_off,
                                                const void* gaddr) {
    asm volatile("global_load_async_to_lds_b128 %0, %1, off"
                 :: "v"(lds_off),
                    "v"((unsigned long long)(uintptr_t)gaddr)
                 : "memory");
}
// Drain all but the newest 16 async ops (in-order retirement).
__device__ __forceinline__ void wait_async_16() {
    asm volatile("s_wait_asynccnt 0x10" ::: "memory");
}

__global__ void cast_f32_bf16_kernel(const float* __restrict__ src,
                                     unsigned short* __restrict__ dst, int n) {
    int i = blockIdx.x * blockDim.x + threadIdx.x;
    if (i < n) dst[i] = f32_to_bf16(src[i]);
}

// Build Wcat[4H][2048] bf16 = [Wx | Wh] row-major (this IS B^T for the WMMA)
__global__ void pack_w_kernel(const float* __restrict__ Wx,
                              const float* __restrict__ Wh,
                              unsigned short* __restrict__ Wcat) {
    int i   = blockIdx.x * blockDim.x + threadIdx.x;   // over 4H*2048
    int row = i >> 11;
    int k   = i & 2047;
    float v = (k < D_) ? Wx[row * D_ + k] : Wh[row * H_ + (k - D_)];
    Wcat[i] = f32_to_bf16(v);
}

__global__ void bias_sum_kernel(const float* __restrict__ bx,
                                const float* __restrict__ bh,
                                float* __restrict__ bs) {
    int i = blockIdx.x * blockDim.x + threadIdx.x;
    if (i < 4 * H_) bs[i] = bx[i] + bh[i];
}

__global__ void init_state_kernel(unsigned int* __restrict__ hbuf32,
                                  unsigned int* __restrict__ bar) {
    int i = blockIdx.x * blockDim.x + threadIdx.x;     // 32768 threads
    if (i < 2 * B_ * H_ / 2) hbuf32[i] = 0;            // 2*B*H bf16 = 32768 u32
    if (i == 0)              *bar      = 0;
}

// Persistent per-layer LSTM kernel. Grid = 64 WGs; each owns 16 h-columns.
// 8 waves: wave w computes gate (w&3) over K-slice (w>>2)*1024..+1024 with
// weights register-resident (32 x v16bf = 256 VGPRs, preloaded once).
// A-operands staged in LDS via async copies: h_{t-1} -> sH (critical path),
// x_{t+1} -> sX[next] prefetched under the current step's GEMM.
__global__ __launch_bounds__(NTH) void lstm_layer_kernel(
    const unsigned short* __restrict__ inp,   // bf16 input sequence
    long long t_str, long long b_str,         // element strides for (t, b)
    const unsigned short* __restrict__ Wcat,  // bf16 [4H][2048]
    const float* __restrict__ bias,           // f32 [4H] = bx+bh
    unsigned short* __restrict__ hbuf,        // bf16 [2][B*H] double buffer
    unsigned short* __restrict__ out_bf16,    // bf16 [S][B*H] (layer0 history)
    float* __restrict__ dout,                 // outputs | h_n | c_n
    int is_last, int layer,
    unsigned int* __restrict__ bar)
{
    extern __shared__ char smem[];
    float* gbuf = (float*)(smem + GBUF_OFF);           // [2][4][32][16]

    const int tid   = threadIdx.x;
    const int lane  = tid & 31;
    const int wave  = tid >> 5;
    const int g     = wave & 3;          // gate id (f,i,g,o)
    const int khalf = wave >> 2;         // K slice id: 0 = x half, 1 = h half
    const int lh    = (lane >> 4) & 1;   // lane-half
    const int nloc  = lane & 15;
    const int hcol_base = blockIdx.x * 16;

    // ---- preload this wave's B-fragments (weights) into registers ----
    const unsigned short* Wrow =
        Wcat + ((size_t)(g * H_ + hcol_base + nloc)) * KTOT
             + khalf * KHALF + lh * 8;
    v16bf bw[KHALF / 32];
    #pragma unroll
    for (int j = 0; j < KHALF / 32; ++j) {
        v8bf blo = *reinterpret_cast<const v8bf*>(Wrow + j * 32);
        v8bf bhi = *reinterpret_cast<const v8bf*>(Wrow + j * 32 + 16);
        bw[j] = __builtin_shufflevector(blo, bhi,
            0,1,2,3,4,5,6,7,8,9,10,11,12,13,14,15);
    }

    const int m0  = nloc;        // A rows for M-tile 0 (b = 0..15)
    const int m1  = nloc + 16;   // A rows for M-tile 1 (b = 16..31)
    const int kh8 = lh * 8;

    // ---- register-resident cell state: thread owns 2 fixed (b,h) cells ----
    float creg[2] = {0.0f, 0.0f};

    // ---- prologue: async-stage x_0 into sX[0] (16 ops/wave) ----
    for (int idx = tid; idx < (B_ * KHALF) / 8; idx += NTH) {
        int b  = idx >> 7;                 // 128 chunks per 1024-elem row
        int ko = (idx & 127) << 3;
        async_load_b128(SX0_OFF + (unsigned)(b * XPITCH + ko) * 2,
                        inp + (size_t)b * b_str + ko);
    }

    for (int t = 0; t < S_; ++t) {
        const unsigned short* hread = hbuf + (size_t)(t & 1) * (B_ * H_);
        unsigned short* hwrite      = hbuf + (size_t)((t + 1) & 1) * (B_ * H_);
        const unsigned sx_cur = (t & 1) ? SX1_OFF : SX0_OFF;
        const unsigned sx_nxt = (t & 1) ? SX0_OFF : SX1_OFF;

        // ---- async-stage h_{t-1} -> sH (needed now; 16 ops/wave) ----
        for (int idx = tid; idx < (B_ * KHALF) / 8; idx += NTH) {
            int b  = idx >> 7;
            int ko = (idx & 127) << 3;
            async_load_b128(SH_OFF + (unsigned)(b * XPITCH + ko) * 2,
                            hread + (size_t)b * H_ + ko);
        }
        // ---- async-prefetch x_{t+1} -> sX[next] (overlaps this GEMM) ----
        {
            int tn = (t + 1 < S_) ? (t + 1) : (S_ - 1);
            for (int idx = tid; idx < (B_ * KHALF) / 8; idx += NTH) {
                int b  = idx >> 7;
                int ko = (idx & 127) << 3;
                async_load_b128(sx_nxt + (unsigned)(b * XPITCH + ko) * 2,
                                inp + (size_t)tn * t_str + (size_t)b * b_str + ko);
            }
        }
        // Drain all but the newest 16 ops: x_t (older) and sH are complete,
        // x_{t+1} may remain in flight.
        wait_async_16();
        __syncthreads();

        // ---- bf16 WMMA GEMM: partial gates[32 x 16], K-slice of 1024 ----
        const unsigned short* Abase =
            (const unsigned short*)(smem + (khalf ? SH_OFF : sx_cur));
        v8f acc0 = {};
        v8f acc1 = {};
        #pragma unroll
        for (int j = 0; j < KHALF / 32; ++j) {
            const int kb = j * 32;
            v8bf a0 = *reinterpret_cast<const v8bf*>(&Abase[m0 * XPITCH + kb + kh8]);
            v8bf a1 = *reinterpret_cast<const v8bf*>(&Abase[m0 * XPITCH + kb + 16 + kh8]);
            v16bf af0 = __builtin_shufflevector(a0, a1,
                0,1,2,3,4,5,6,7,8,9,10,11,12,13,14,15);

            v8bf a2 = *reinterpret_cast<const v8bf*>(&Abase[m1 * XPITCH + kb + kh8]);
            v8bf a3 = *reinterpret_cast<const v8bf*>(&Abase[m1 * XPITCH + kb + 16 + kh8]);
            v16bf af1 = __builtin_shufflevector(a2, a3,
                0,1,2,3,4,5,6,7,8,9,10,11,12,13,14,15);

            acc0 = __builtin_amdgcn_wmma_f32_16x16x32_bf16(
                false, af0, false, bw[j], (short)0, acc0, false, false);
            acc1 = __builtin_amdgcn_wmma_f32_16x16x32_bf16(
                false, af1, false, bw[j], (short)0, acc1, false, false);
        }

        // ---- exchange gate tiles through LDS (C/D layout: m = r + 8*lh) ----
        #pragma unroll
        for (int r = 0; r < 8; ++r) {
            gbuf[((khalf * 4 + g) * 32 +      lh * 8 + r) * 16 + nloc] = acc0[r];
            gbuf[((khalf * 4 + g) * 32 + 16 + lh * 8 + r) * 16 + nloc] = acc1[r];
        }
        __syncthreads();

        // ---- LSTM cell: 2 fixed cells per thread, c in registers ----
        #pragma unroll
        for (int q = 0; q < 2; ++q) {
            int i = tid + q * NTH;       // 0..511 over [32 x 16]
            int b = i >> 4;
            int n = i & 15;
            int hcol = hcol_base + n;
            float fg = gbuf[((0 * 4 + 0) * 32 + b) * 16 + n]
                     + gbuf[((1 * 4 + 0) * 32 + b) * 16 + n] + bias[0 * H_ + hcol];
            float ig = gbuf[((0 * 4 + 1) * 32 + b) * 16 + n]
                     + gbuf[((1 * 4 + 1) * 32 + b) * 16 + n] + bias[1 * H_ + hcol];
            float gg = gbuf[((0 * 4 + 2) * 32 + b) * 16 + n]
                     + gbuf[((1 * 4 + 2) * 32 + b) * 16 + n] + bias[2 * H_ + hcol];
            float og = gbuf[((0 * 4 + 3) * 32 + b) * 16 + n]
                     + gbuf[((1 * 4 + 3) * 32 + b) * 16 + n] + bias[3 * H_ + hcol];
            float sf = 1.0f / (1.0f + __expf(-fg));
            float si = 1.0f / (1.0f + __expf(-ig));
            float so = 1.0f / (1.0f + __expf(-og));
            float c_new = sf * creg[q] + si * tanhf(gg);
            float h_new = so * tanhf(c_new);
            creg[q] = c_new;
            hwrite[(size_t)b * H_ + hcol] = f32_to_bf16(h_new);
            if (is_last) {
                dout[(size_t)b * (S_ * (size_t)H_) + (size_t)t * H_ + hcol] = h_new;
            } else {
                out_bf16[(size_t)t * (B_ * H_) + (size_t)b * H_ + hcol] =
                    f32_to_bf16(h_new);
            }
            if (t == S_ - 1) {
                size_t base = (size_t)B_ * S_ * H_;
                dout[base + (size_t)layer * (B_ * H_)
                          + (size_t)b * H_ + hcol] = h_new;                 // h_n
                dout[base + (size_t)2 * (B_ * H_) + (size_t)layer * (B_ * H_)
                          + (size_t)b * H_ + hcol] = c_new;                 // c_n
            }
        }

        // ---- device-wide step barrier (64 WGs, monotonic counter) ----
        __threadfence();
        __syncthreads();
        if (tid == 0) {
            __hip_atomic_fetch_add(bar, 1u, __ATOMIC_RELEASE,
                                   __HIP_MEMORY_SCOPE_AGENT);
            unsigned int target = (unsigned int)(t + 1) * (unsigned int)gridDim.x;
            while (__hip_atomic_load(bar, __ATOMIC_ACQUIRE,
                                     __HIP_MEMORY_SCOPE_AGENT) < target) {
                __builtin_amdgcn_s_sleep(2);
            }
        }
        __syncthreads();
    }
}

extern "C" void kernel_launch(void* const* d_in, const int* in_sizes, int n_in,
                              void* d_out, int out_size, void* d_ws, size_t ws_size,
                              hipStream_t stream) {
    (void)in_sizes; (void)n_in; (void)out_size; (void)ws_size;
    const float* x   = (const float*)d_in[0];
    const float* Wx0 = (const float*)d_in[1];
    const float* Wh0 = (const float*)d_in[2];
    const float* bx0 = (const float*)d_in[3];
    const float* bh0 = (const float*)d_in[4];
    const float* Wx1 = (const float*)d_in[5];
    const float* Wh1 = (const float*)d_in[6];
    const float* bx1 = (const float*)d_in[7];
    const float* bh1 = (const float*)d_in[8];
    float* out = (float*)d_out;

    char* ws = (char*)d_ws;
    size_t off = 0;
    auto alloc = [&](size_t bytes) -> char* {
        char* p = ws + off;
        off += (bytes + 255) & ~(size_t)255;
        return p;
    };
    unsigned short* x_bf   = (unsigned short*)alloc((size_t)B_ * S_ * D_ * 2); // 32 MB
    unsigned short* w0     = (unsigned short*)alloc((size_t)4 * H_ * KTOT * 2);// 16 MB
    unsigned short* w1     = (unsigned short*)alloc((size_t)4 * H_ * KTOT * 2);// 16 MB
    unsigned short* h0_all = (unsigned short*)alloc((size_t)S_ * B_ * H_ * 2); // 32 MB
    float*          b0     = (float*)alloc((size_t)4 * H_ * 4);
    float*          b1     = (float*)alloc((size_t)4 * H_ * 4);
    unsigned short* hb0    = (unsigned short*)alloc((size_t)2 * B_ * H_ * 2);
    unsigned short* hb1    = (unsigned short*)alloc((size_t)2 * B_ * H_ * 2);
    unsigned int*   bar0   = (unsigned int*)alloc(256);
    unsigned int*   bar1   = (unsigned int*)alloc(256);

    { // x -> bf16
        int n = B_ * S_ * D_;
        cast_f32_bf16_kernel<<<(n + 255) / 256, 256, 0, stream>>>(x, x_bf, n);
    }
    pack_w_kernel<<<(4 * H_ * KTOT) / 256, 256, 0, stream>>>(Wx0, Wh0, w0);
    pack_w_kernel<<<(4 * H_ * KTOT) / 256, 256, 0, stream>>>(Wx1, Wh1, w1);
    bias_sum_kernel<<<(4 * H_ + 255) / 256, 256, 0, stream>>>(bx0, bh0, b0);
    bias_sum_kernel<<<(4 * H_ + 255) / 256, 256, 0, stream>>>(bx1, bh1, b1);

    size_t shmem = SHMEM_B;

    // Layer 0: input = x [B,S,D] -> strides (t: D, b: S*D)
    init_state_kernel<<<128, 256, 0, stream>>>((unsigned int*)hb0, bar0);
    lstm_layer_kernel<<<NWG, NTH, shmem, stream>>>(
        x_bf, (long long)D_, (long long)S_ * D_, w0, b0, hb0,
        h0_all, out, /*is_last=*/0, /*layer=*/0, bar0);

    // Layer 1: input = h0_all [S,B,H] -> strides (t: B*H, b: H)
    init_state_kernel<<<128, 256, 0, stream>>>((unsigned int*)hb1, bar1);
    lstm_layer_kernel<<<NWG, NTH, shmem, stream>>>(
        h0_all, (long long)B_ * H_, (long long)H_, w1, b1, hb1,
        h0_all /*unused*/, out, /*is_last=*/1, /*layer=*/1, bar1);
}
